// VanillaMultiHeadAttention_8761733284608
// MI455X (gfx1250) — matrix operand order repaired and checked by reading, not compiled, not run on hardware
//
#include <hip/hip_runtime.h>

typedef __attribute__((ext_vector_type(16))) __bf16 v16bf;
typedef __attribute__((ext_vector_type(8)))  float  v8f;

#define C_DIM 1024
#define T_DIM 1024
#define B_DIM 2
#define H_DIM 16
#define D_DIM 64
#define NROWS (B_DIM * T_DIM)   // 2048

__device__ __forceinline__ unsigned short f2bf(float f) {
  union { __bf16 b; unsigned short u; } cv;
  cv.b = (__bf16)f;
  return cv.u;
}

__device__ __forceinline__ v8f zero8() {
  v8f z = {0.f, 0.f, 0.f, 0.f, 0.f, 0.f, 0.f, 0.f};
  return z;
}

__device__ __forceinline__ v8f wmma_bf16(v16bf a, v16bf b, v8f c) {
  // D = A(16x32) * B(32x16) + C, f32 accumulate
  return __builtin_amdgcn_wmma_f32_16x16x32_bf16(false, a, false, b, (short)0, c,
                                                 false, false);
}

// A-operand fragment: 16(M) x 32(K) bf16 tile from row-major memory, ld elems.
// Lane m holds K = half*8 + [0..7] and 16 + half*8 + [0..7]  (ISA A layout).
__device__ __forceinline__ v16bf load_fragA(const unsigned short* p, int ld, int lane) {
  const int m = lane & 15, hf = lane >> 4;
  const unsigned short* q = p + m * ld + hf * 8;
  union { v16bf v; uint4 u[2]; } r;
  r.u[0] = *(const uint4*)(q);
  r.u[1] = *(const uint4*)(q + 16);
  return r.v;
}

// B-operand fragment: 32(K) x 16(N) tile, sourced from a row-major N x K
// matrix (row = output column n). Lane n holds K = half*8 + [0..7] and 16 + half*8 + [0..7].
__device__ __forceinline__ v16bf load_fragB(const unsigned short* p, int ld, int lane) {
  const int n = lane & 15, hf = lane >> 4;
  const unsigned short* q = p + n * ld + hf * 8;
  union { v16bf v; uint4 u[2]; } r;
  r.u[0] = *(const uint4*)(q);
  r.u[1] = *(const uint4*)(q + 16);
  return r.v;
}

// ---------------------------------------------------------------- convert ---
__global__ void cvt_f32_bf16(const float* __restrict__ src,
                             unsigned short* __restrict__ dst, int n) {
  int i = blockIdx.x * blockDim.x + threadIdx.x;
  const int stride = gridDim.x * blockDim.x;
  for (; i < n; i += stride) dst[i] = f2bf(src[i]);
}

// ------------------------------------------------------- projection GEMMs ---
// Out = X(2048x1024) * Wk^T + bk for X in {q,k,v}; gridDim.y selects input.
// One wave computes a 32x32 output tile (2x2 WMMA register block: 4 WMMAs per
// 8 b128 loads). V's projection (y==2) is stored transposed as [B,H,D,T] so
// the attention PV B-fragments become contiguous b128 loads.
__global__ __launch_bounds__(256) void gemm_bf16_qkv(
    const unsigned short* __restrict__ X0, const unsigned short* __restrict__ X1,
    const unsigned short* __restrict__ X2, const unsigned short* __restrict__ W,
    const float* __restrict__ bias, unsigned short* __restrict__ O0,
    unsigned short* __restrict__ O1, unsigned short* __restrict__ O2) {
  const int lane = threadIdx.x & 31;
  const int tile = blockIdx.x * 8 + (threadIdx.x >> 5);  // 2048 tiles of 32x32
  const int R = (tile >> 5) << 5;                        // 64 row-tiles
  const int col0 = (tile & 31) << 5;                     // 32 col-tiles
  const int which = blockIdx.y;
  const unsigned short* X = (which == 0) ? X0 : (which == 1) ? X1 : X2;
  unsigned short* Out = (which == 0) ? O0 : (which == 1) ? O1 : O2;
  const int transposed = (which == 2);

  v8f a00 = zero8(), a01 = zero8(), a10 = zero8(), a11 = zero8();
  for (int k0 = 0; k0 < C_DIM; k0 += 32) {
    __builtin_prefetch(X + R * C_DIM + k0 + 64, 0, 1);          // next K-block
    __builtin_prefetch(W + col0 * C_DIM + k0 + 64, 0, 1);
    v16bf fa0 = load_fragA(X + R * C_DIM + k0, C_DIM, lane);
    v16bf fa1 = load_fragA(X + (R + 16) * C_DIM + k0, C_DIM, lane);
    v16bf fb0 = load_fragB(W + col0 * C_DIM + k0, C_DIM, lane);
    v16bf fb1 = load_fragB(W + (col0 + 16) * C_DIM + k0, C_DIM, lane);
    a00 = wmma_bf16(fa0, fb0, a00);
    a01 = wmma_bf16(fa0, fb1, a01);
    a10 = wmma_bf16(fa1, fb0, a10);
    a11 = wmma_bf16(fa1, fb1, a11);
  }

  const int hf = lane >> 4, n = lane & 15;
  const float bv0 = bias[col0 + n];
  const float bv1 = bias[col0 + 16 + n];
#pragma unroll
  for (int j = 0; j < 8; ++j) {
    const float vals[4] = {a00[j] + bv0, a01[j] + bv1, a10[j] + bv0, a11[j] + bv1};
#pragma unroll
    for (int t = 0; t < 4; ++t) {
      const int row = R + (t >> 1) * 16 + j + 8 * hf;   // global row [0,2048)
      const int ch = col0 + (t & 1) * 16 + n;           // h*64 + d
      if (!transposed) {
        Out[row * C_DIM + ch] = f2bf(vals[t]);
      } else {
        const int bb = row >> 10, tok = row & (T_DIM - 1);
        Out[((bb * H_DIM + (ch >> 6)) * D_DIM + (ch & 63)) * T_DIM + tok] =
            f2bf(vals[t]);
      }
    }
  }
}

// -------------------------------------------------------- flash attention ---
__global__ __launch_bounds__(256) void flash_attn(
    const unsigned short* __restrict__ qp,   // [B*T, C]
    const unsigned short* __restrict__ kp,   // [B*T, C]
    const unsigned short* __restrict__ vpT,  // [B*H*D, T]
    unsigned short* __restrict__ y) {        // [B*T, C]
  __shared__ unsigned short pbuf[8][16 * 32];  // per-wave P staging tile
  const int lane = threadIdx.x & 31;
  const int wv = threadIdx.x >> 5;
  const int w = blockIdx.x * 8 + wv;          // 2048 waves total
  const int b = w >> 10;
  const int h = (w >> 6) & (H_DIM - 1);
  const int R = (w & 63) << 4;                // query tile start
  const int hf = lane >> 4, n15 = lane & 15;

  const unsigned short* qbase = qp + (b * T_DIM + R) * C_DIM + h * D_DIM;
  const v16bf aq0 = load_fragA(qbase, C_DIM, lane);
  const v16bf aq1 = load_fragA(qbase + 32, C_DIM, lane);

  v8f o0 = zero8(), o1 = zero8(), o2 = zero8(), o3 = zero8();
  float m_s[8], l_s[8];
#pragma unroll
  for (int j = 0; j < 8; ++j) { m_s[j] = -3.0e38f; l_s[j] = 0.f; }

  for (int kk = 0; kk < R + 16; kk += 32) {
    // S = Q K^T for 16 queries x 32 keys (two 16-wide N halves)
    const unsigned short* kb0 = kp + (b * T_DIM + kk) * C_DIM + h * D_DIM;
    const unsigned short* kb1 = kb0 + 16 * C_DIM;
    v8f s0 = wmma_bf16(aq1, load_fragB(kb0 + 32, C_DIM, lane),
             wmma_bf16(aq0, load_fragB(kb0,      C_DIM, lane), zero8()));
    v8f s1 = wmma_bf16(aq1, load_fragB(kb1 + 32, C_DIM, lane),
             wmma_bf16(aq0, load_fragB(kb1,      C_DIM, lane), zero8()));

#pragma unroll
    for (int j = 0; j < 8; ++j) {
      const int row = R + j + 8 * hf;
      const float v0 = (kk + n15      <= row) ? s0[j] * 0.125f : -3.0e38f;
      const float v1 = (kk + 16 + n15 <= row) ? s1[j] * 0.125f : -3.0e38f;
      // row-max across the 16 N-lanes of this half-wave
      float t = fmaxf(v0, v1);
      t = fmaxf(t, __shfl_xor(t, 1, 32));
      t = fmaxf(t, __shfl_xor(t, 2, 32));
      t = fmaxf(t, __shfl_xor(t, 4, 32));
      t = fmaxf(t, __shfl_xor(t, 8, 32));
      const float mnew  = fmaxf(m_s[j], t);
      const float alpha = __expf(m_s[j] - mnew);
      const float p0 = __expf(v0 - mnew);
      const float p1 = __expf(v1 - mnew);
      float r = p0 + p1;
      r += __shfl_xor(r, 1, 32);
      r += __shfl_xor(r, 2, 32);
      r += __shfl_xor(r, 4, 32);
      r += __shfl_xor(r, 8, 32);
      l_s[j] = l_s[j] * alpha + r;
      m_s[j] = mnew;
      o0[j] *= alpha; o1[j] *= alpha; o2[j] *= alpha; o3[j] *= alpha;
      // C-layout -> LDS (row-major 16x32 P tile)
      pbuf[wv][(j + 8 * hf) * 32 + n15]      = f2bf(p0);
      pbuf[wv][(j + 8 * hf) * 32 + 16 + n15] = f2bf(p1);
    }
    asm volatile("s_wait_dscnt 0" ::: "memory");
    const v16bf pf = load_fragA(&pbuf[wv][0], 32, lane);  // A-layout reload

    // O += P(16x32) * V(32x64): vpT rows are d, cols are tokens
    const unsigned short* vb = vpT + (b * H_DIM + h) * D_DIM * T_DIM + kk;
    o0 = wmma_bf16(pf, load_fragB(vb,              T_DIM, lane), o0);
    o1 = wmma_bf16(pf, load_fragB(vb + 16 * T_DIM, T_DIM, lane), o1);
    o2 = wmma_bf16(pf, load_fragB(vb + 32 * T_DIM, T_DIM, lane), o2);
    o3 = wmma_bf16(pf, load_fragB(vb + 48 * T_DIM, T_DIM, lane), o3);
  }

#pragma unroll
  for (int j = 0; j < 8; ++j) {
    const float inv = 1.0f / l_s[j];
    unsigned short* yr =
        y + (b * T_DIM + R + j + 8 * hf) * C_DIM + h * D_DIM + n15;
    yr[0]  = f2bf(o0[j] * inv);
    yr[16] = f2bf(o1[j] * inv);
    yr[32] = f2bf(o2[j] * inv);
    yr[48] = f2bf(o3[j] * inv);
  }
}

// ------------------------------------------------------- output projection --
// d_out = Y(2048x1024) * Wc^T + bc, f32 output. 32x32 tile per wave.
__global__ __launch_bounds__(256) void gemm_bf16_out(
    const unsigned short* __restrict__ Y, const unsigned short* __restrict__ W,
    const float* __restrict__ bias, float* __restrict__ Out) {
  const int lane = threadIdx.x & 31;
  const int tile = blockIdx.x * 8 + (threadIdx.x >> 5);  // 2048 tiles of 32x32
  const int R = (tile >> 5) << 5;
  const int col0 = (tile & 31) << 5;

  v8f a00 = zero8(), a01 = zero8(), a10 = zero8(), a11 = zero8();
  for (int k0 = 0; k0 < C_DIM; k0 += 32) {
    __builtin_prefetch(Y + R * C_DIM + k0 + 64, 0, 1);
    __builtin_prefetch(W + col0 * C_DIM + k0 + 64, 0, 1);
    v16bf fa0 = load_fragA(Y + R * C_DIM + k0, C_DIM, lane);
    v16bf fa1 = load_fragA(Y + (R + 16) * C_DIM + k0, C_DIM, lane);
    v16bf fb0 = load_fragB(W + col0 * C_DIM + k0, C_DIM, lane);
    v16bf fb1 = load_fragB(W + (col0 + 16) * C_DIM + k0, C_DIM, lane);
    a00 = wmma_bf16(fa0, fb0, a00);
    a01 = wmma_bf16(fa0, fb1, a01);
    a10 = wmma_bf16(fa1, fb0, a10);
    a11 = wmma_bf16(fa1, fb1, a11);
  }

  const int hf = lane >> 4, n = lane & 15;
  const float bv0 = bias[col0 + n];
  const float bv1 = bias[col0 + 16 + n];
#pragma unroll
  for (int j = 0; j < 8; ++j) {
    const int r0 = R + j + 8 * hf;
    Out[r0 * C_DIM + col0 + n]             = a00[j] + bv0;
    Out[r0 * C_DIM + col0 + 16 + n]        = a01[j] + bv1;
    Out[(r0 + 16) * C_DIM + col0 + n]      = a10[j] + bv0;
    Out[(r0 + 16) * C_DIM + col0 + 16 + n] = a11[j] + bv1;
  }
}

// ------------------------------------------------------------------ launch --
extern "C" void kernel_launch(void* const* d_in, const int* in_sizes, int n_in,
                              void* d_out, int out_size, void* d_ws, size_t ws_size,
                              hipStream_t stream) {
  (void)in_sizes; (void)n_in; (void)out_size; (void)ws_size;
  const float* q  = (const float*)d_in[0];
  const float* k  = (const float*)d_in[1];
  const float* v  = (const float*)d_in[2];
  // d_in[3] = boolean causal mask (recomputed analytically; unused)
  const float* Wk = (const float*)d_in[4];
  const float* bk = (const float*)d_in[5];
  const float* Wc = (const float*)d_in[6];
  const float* bc = (const float*)d_in[7];

  char* ws = (char*)d_ws;
  const size_t MB = 1024 * 1024;
  unsigned short* qb  = (unsigned short*)(ws + 0 * MB);   // 4 MB each
  unsigned short* kb  = (unsigned short*)(ws + 4 * MB);
  unsigned short* vb  = (unsigned short*)(ws + 8 * MB);
  unsigned short* Wkb = (unsigned short*)(ws + 12 * MB);  // 2 MB
  unsigned short* Wcb = (unsigned short*)(ws + 14 * MB);  // 2 MB
  unsigned short* qp  = (unsigned short*)(ws + 16 * MB);
  unsigned short* kp  = (unsigned short*)(ws + 20 * MB);
  unsigned short* vpT = (unsigned short*)(ws + 24 * MB);
  unsigned short* y   = (unsigned short*)(ws + 28 * MB);

  const int nBTC = B_DIM * T_DIM * C_DIM;  // 2M
  const int nCC  = C_DIM * C_DIM;          // 1M

  cvt_f32_bf16<<<2048, 256, 0, stream>>>(q,  qb,  nBTC);
  cvt_f32_bf16<<<2048, 256, 0, stream>>>(k,  kb,  nBTC);
  cvt_f32_bf16<<<2048, 256, 0, stream>>>(v,  vb,  nBTC);
  cvt_f32_bf16<<<1024, 256, 0, stream>>>(Wk, Wkb, nCC);
  cvt_f32_bf16<<<1024, 256, 0, stream>>>(Wc, Wcb, nCC);

  // q/k/v projections in one launch: gridDim = (256 blocks, 3 inputs)
  gemm_bf16_qkv<<<dim3(256, 3), 256, 0, stream>>>(qb, kb, vb, Wkb, bk,
                                                  qp, kp, vpT);

  flash_attn<<<256, 256, 0, stream>>>(qp, kp, vpT, y);

  gemm_bf16_out<<<256, 256, 0, stream>>>(y, Wcb, bc, (float*)d_out);
}
